// Encoder_89893665505565
// MI455X (gfx1250) — matrix operand order repaired and checked by reading, not compiled, not run on hardware
//
#include <hip/hip_runtime.h>

typedef __bf16 bf16_t;
typedef __attribute__((ext_vector_type(16))) __bf16 bf16x16;
typedef __attribute__((ext_vector_type(8)))  float  f32x8;

union FragAB { bf16x16 v; uint4 q[2]; };

// ---------------------------------------------------------------------------
// CDNA5 async global->LDS copy (16B per lane), ASYNCcnt-tracked.
// Builtin signature (from hipcc diagnostic): param1 = v4i in AS(1) (global),
// param2 = v4i in AS(3) (LDS), then imm offset + cpol.
// ---------------------------------------------------------------------------
typedef int v4i __attribute__((vector_size(16)));
typedef __attribute__((address_space(1))) v4i* gas1_t;
typedef __attribute__((address_space(3))) v4i* las3_t;

__device__ __forceinline__ void async_ld16(const bf16_t* g, bf16_t* l) {
#if __has_builtin(__builtin_amdgcn_global_load_async_to_lds_b128)
  __builtin_amdgcn_global_load_async_to_lds_b128(
      (gas1_t)(uintptr_t)g,
      (las3_t)(unsigned)(uintptr_t)l, 0, 0);
#else
  asm volatile("global_load_async_to_lds_b128 %0, %1, off"
               :: "v"((unsigned)(uintptr_t)l),
                  "v"((unsigned long long)(uintptr_t)g)
               : "memory");
#endif
}

__device__ __forceinline__ void wait_async_le4() {
#if __has_builtin(__builtin_amdgcn_s_wait_asynccnt)
  __builtin_amdgcn_s_wait_asynccnt(4);
#else
  asm volatile("s_wait_asynccnt 0x4" ::: "memory");
#endif
}

__device__ __forceinline__ void wait_async_le0() {
#if __has_builtin(__builtin_amdgcn_s_wait_asynccnt)
  __builtin_amdgcn_s_wait_asynccnt(0);
#else
  asm volatile("s_wait_asynccnt 0x0" ::: "memory");
#endif
}

// ---------------------------------------------------------------------------
// Elementwise / reduction kernels
// ---------------------------------------------------------------------------

// Wt[n][k] = bf16(W[k][n]);  W is [K][N] row-major, Wt is [N][K] row-major.
__global__ __launch_bounds__(256) void enc_cvt_transpose(
    const float* __restrict__ W, bf16_t* __restrict__ Wt, int K, int N) {
  int idx = blockIdx.x * 256 + threadIdx.x;
  int k = idx % K;
  int n = idx / K;
  Wt[idx] = (bf16_t)W[(size_t)k * N + n];
}

// vt[b][dv][t] = qkv[b][t][2048 + dv]   (V slice of qkv, transposed per batch)
__global__ __launch_bounds__(256) void enc_vtranspose(
    const bf16_t* __restrict__ qkv, bf16_t* __restrict__ vt) {
  int idx = blockIdx.x * 256 + threadIdx.x;   // flat over [4][1024][2048]
  int t  = idx & 2047;
  int dv = (idx >> 11) & 1023;
  int b  = idx >> 21;
  vt[idx] = qkv[((size_t)(b * 2048 + t)) * 3072 + 2048 + dv];
}

// Row LayerNorm over d=1024, writes bf16 normalized output.
__global__ __launch_bounds__(256) void enc_layernorm(
    const float* __restrict__ x, const float* __restrict__ g,
    const float* __restrict__ be, bf16_t* __restrict__ out) {
  __shared__ float red[256];
  const int tid = threadIdx.x;
  const size_t row = blockIdx.x;
  const float* xr = x + row * 1024;
  float v[4];
  float s = 0.f;
#pragma unroll
  for (int i = 0; i < 4; ++i) { v[i] = xr[tid + i * 256]; s += v[i]; }
  red[tid] = s; __syncthreads();
  for (int o = 128; o > 0; o >>= 1) { if (tid < o) red[tid] += red[tid + o]; __syncthreads(); }
  const float mu = red[0] * (1.f / 1024.f);
  __syncthreads();
  float sq = 0.f;
#pragma unroll
  for (int i = 0; i < 4; ++i) { float d = v[i] - mu; sq += d * d; }
  red[tid] = sq; __syncthreads();
  for (int o = 128; o > 0; o >>= 1) { if (tid < o) red[tid] += red[tid + o]; __syncthreads(); }
  const float r = rsqrtf(red[0] * (1.f / 1024.f) + 1e-5f);
#pragma unroll
  for (int i = 0; i < 4; ++i) {
    int c = tid + i * 256;
    out[row * 1024 + c] = (bf16_t)((v[i] - mu) * r * g[c] + be[c]);
  }
}

// Row softmax over 2048 f32 scores -> bf16 probabilities.
__global__ __launch_bounds__(256) void enc_softmax(
    const float* __restrict__ S, bf16_t* __restrict__ P) {
  __shared__ float red[256];
  const int tid = threadIdx.x;
  const size_t row = blockIdx.x;
  const float* sr = S + row * 2048;
  float e[8];
  float m = -1e30f;
#pragma unroll
  for (int i = 0; i < 8; ++i) { e[i] = sr[tid + i * 256]; m = fmaxf(m, e[i]); }
  red[tid] = m; __syncthreads();
  for (int o = 128; o > 0; o >>= 1) { if (tid < o) red[tid] = fmaxf(red[tid], red[tid + o]); __syncthreads(); }
  m = red[0];
  __syncthreads();
  float s = 0.f;
#pragma unroll
  for (int i = 0; i < 8; ++i) { e[i] = __expf(e[i] - m); s += e[i]; }
  red[tid] = s; __syncthreads();
  for (int o = 128; o > 0; o >>= 1) { if (tid < o) red[tid] += red[tid + o]; __syncthreads(); }
  const float inv = 1.f / red[0];
  bf16_t* pr = P + row * 2048;
#pragma unroll
  for (int i = 0; i < 8; ++i) pr[tid + i * 256] = (bf16_t)(e[i] * inv);
}

// ---------------------------------------------------------------------------
// Generic bf16 WMMA GEMM:  C[M,N] = A[M,K] @ Bt[N,K]^T (+ epilogue)
//   MODE 0: bf16 out = acc + bias            (QKV proj, attn@V with bias=null)
//   MODE 1: f32  out = acc * scale           (attention scores)
//   MODE 2: f32  out = acc + bias + residual (O-proj / FF2 with residual add)
//   MODE 3: bf16 out = gelu(acc + bias)      (FF1, exact erf GELU)
// Block: 256 threads = 8 wave32, tile 128x128, BK=32.
// Double-buffered LDS fed by GLOBAL_LOAD_ASYNC_TO_LDS_B128 (ASYNCcnt pipeline):
// tile s+1 streams into LDS while WMMAs consume tile s.
// Assumes M%128==0, N%128==0, K%32==0 (true for all shapes here).
// ---------------------------------------------------------------------------
template <int MODE>
__global__ __launch_bounds__(256) void enc_gemm(
    const bf16_t* __restrict__ A, int lda, long long strideA,
    const bf16_t* __restrict__ Bt, int ldb, long long strideB,
    void* Cv, int ldc, long long strideC,
    const float* __restrict__ bias,
    const float* res, long long strideRes,
    int K, float scale) {
  __shared__ bf16_t As[2][128][40];  // pad to 40 bf16 (80B) -> conflict-free frags
  __shared__ bf16_t Bs[2][128][40];

  const int tid  = threadIdx.x;
  const int lane = tid & 31;
  const int wave = tid >> 5;
  const int wm = wave >> 1;        // 0..3  (32-row slice)
  const int wn = wave & 1;         // 0..1  (64-col slice)
  const int lm = lane & 15;
  const int kh = lane >> 4;        // K-half select per ISA A/B layouts

  const size_t z = blockIdx.z;
  A  += z * strideA;
  Bt += z * strideB;

  // cooperative-load coordinates: each thread moves 16 bf16 of A and of B
  const int ldRow = tid >> 1;          // 0..127
  const int ldK   = (tid & 1) << 4;    // 0 or 16
  const bf16_t* Ag = A  + (size_t)(blockIdx.y * 128 + ldRow) * lda + ldK;
  const bf16_t* Bg = Bt + (size_t)(blockIdx.x * 128 + ldRow) * ldb + ldK;

  f32x8 acc[2][4];
#pragma unroll
  for (int i = 0; i < 2; ++i)
#pragma unroll
    for (int j = 0; j < 4; ++j)
#pragma unroll
      for (int v = 0; v < 8; ++v) acc[i][j][v] = 0.f;

  // 4 async b128 ops per thread per tile (2 for A, 2 for B)
  auto issueTile = [&](int buf, int k0) {
    async_ld16(Ag + k0,     &As[buf][ldRow][ldK]);
    async_ld16(Ag + k0 + 8, &As[buf][ldRow][ldK + 8]);
    async_ld16(Bg + k0,     &Bs[buf][ldRow][ldK]);
    async_ld16(Bg + k0 + 8, &Bs[buf][ldRow][ldK + 8]);
  };

  const int nsteps = K >> 5;
  issueTile(0, 0);

  for (int s = 0; s < nsteps; ++s) {
    const int cur = s & 1;
    if (s + 1 < nsteps) {
      issueTile(cur ^ 1, (s + 1) << 5);   // prefetch next tile into other buffer
      wait_async_le4();                   // this wave's current-tile ops done
    } else {
      wait_async_le0();
    }
    __syncthreads();                      // all waves' current tile visible

    FragAB fa[2];
#pragma unroll
    for (int i = 0; i < 2; ++i) {
      const int m = wm * 32 + i * 16 + lm;
      // 16-bit A 16x32 layout: lanes hold row m; K = 8*kh+j (j<8), 16+8*kh+j-8 (j>=8)
      fa[i].q[0] = *(const uint4*)&As[cur][m][kh * 8];
      fa[i].q[1] = *(const uint4*)&As[cur][m][16 + kh * 8];
    }
#pragma unroll
    for (int j = 0; j < 4; ++j) {
      const int n = wn * 64 + j * 16 + lm;
      // 16-bit B 32x16 layout: lane = col n; K = 8*kh+j (j<8), 16+8*kh+j-8 (j>=8)
      FragAB fb;
      fb.q[0] = *(const uint4*)&Bs[cur][n][kh * 8];
      fb.q[1] = *(const uint4*)&Bs[cur][n][16 + kh * 8];
#pragma unroll
      for (int i = 0; i < 2; ++i)
        acc[i][j] = __builtin_amdgcn_wmma_f32_16x16x32_bf16(
            false, fa[i].v, false, fb.v, (short)0, acc[i][j], false, false);
    }
    __syncthreads();   // all waves done reading 'cur' before it is refilled
  }

  // epilogue: C layout — VGPR v holds row v + 8*(lane/16), col = lane%16
  const int rBase = blockIdx.y * 128 + wm * 32;
  const int cBase = blockIdx.x * 128 + wn * 64;
  float*  Cf = (float*)Cv  + z * strideC;
  bf16_t* Cb = (bf16_t*)Cv + z * strideC;
  if constexpr (MODE == 2) { res += z * strideRes; }

#pragma unroll
  for (int i = 0; i < 2; ++i)
#pragma unroll
    for (int j = 0; j < 4; ++j) {
      const int col = cBase + j * 16 + lm;
      float bv = 0.f;
      if constexpr (MODE != 1) { if (bias) bv = bias[col]; }
#pragma unroll
      for (int v = 0; v < 8; ++v) {
        const int row = rBase + i * 16 + v + 8 * kh;
        const size_t off = (size_t)row * ldc + col;
        const float a = acc[i][j][v];
        if constexpr (MODE == 0) {
          Cb[off] = (bf16_t)(a + bv);
        } else if constexpr (MODE == 1) {
          Cf[off] = a * scale;
        } else if constexpr (MODE == 2) {
          Cf[off] = a + bv + res[off];
        } else {  // MODE == 3, exact GELU
          const float t = a + bv;
          Cb[off] = (bf16_t)(0.5f * t * (1.0f + erff(t * 0.70710678118654752f)));
        }
      }
    }
}

// ---------------------------------------------------------------------------
// Host launch
// ---------------------------------------------------------------------------
extern "C" void kernel_launch(void* const* d_in, const int* in_sizes, int n_in,
                              void* d_out, int out_size, void* d_ws, size_t ws_size,
                              hipStream_t stream) {
  (void)in_sizes; (void)n_in; (void)out_size; (void)ws_size;
  const float* x      = (const float*)d_in[0];   // [4,2048,1024]
  const float* W_qkv  = (const float*)d_in[1];   // [1024,3072]
  const float* b_qkv  = (const float*)d_in[2];
  const float* W_o    = (const float*)d_in[3];   // [1024,1024]
  const float* b_o    = (const float*)d_in[4];
  const float* W_ff1  = (const float*)d_in[5];   // [1024,4096]
  const float* b_ff1  = (const float*)d_in[6];
  const float* W_ff2  = (const float*)d_in[7];   // [4096,1024]
  const float* b_ff2  = (const float*)d_in[8];
  const float* g1     = (const float*)d_in[9];
  const float* be1    = (const float*)d_in[10];
  const float* g2     = (const float*)d_in[11];
  const float* be2    = (const float*)d_in[12];
  float* out = (float*)d_out;                    // [4,2048,1024]

  // workspace carve-up
  char* wp = (char*)d_ws;
  bf16_t* wqkvT = (bf16_t*)wp; wp += 3072ULL * 1024 * 2;
  bf16_t* woT   = (bf16_t*)wp; wp += 1024ULL * 1024 * 2;
  bf16_t* wff1T = (bf16_t*)wp; wp += 4096ULL * 1024 * 2;
  bf16_t* wff2T = (bf16_t*)wp; wp += 1024ULL * 4096 * 2;
  bf16_t* hbuf  = (bf16_t*)wp; wp += 8192ULL * 1024 * 2;     // LN1 / LN2 output
  bf16_t* qkv   = (bf16_t*)wp; wp += 8192ULL * 3072 * 2;
  bf16_t* vt    = (bf16_t*)wp; wp += 4ULL * 1024 * 2048 * 2;
  bf16_t* attno = (bf16_t*)wp; wp += 8192ULL * 1024 * 2;
  float*  scoresF = (float*)wp;
  bf16_t* ff1     = (bf16_t*)wp;                 // reuses score region (post-attn)
  wp += 4ULL * 2048 * 2048 * 4;
  bf16_t* P = (bf16_t*)wp; wp += 4ULL * 2048 * 2048 * 2;

  const dim3 blk(256);

  // 1) weights -> bf16, transposed to [N][K]
  enc_cvt_transpose<<<(1024 * 3072) / 256, blk, 0, stream>>>(W_qkv, wqkvT, 1024, 3072);
  enc_cvt_transpose<<<(1024 * 1024) / 256, blk, 0, stream>>>(W_o,   woT,   1024, 1024);
  enc_cvt_transpose<<<(1024 * 4096) / 256, blk, 0, stream>>>(W_ff1, wff1T, 1024, 4096);
  enc_cvt_transpose<<<(4096 * 1024) / 256, blk, 0, stream>>>(W_ff2, wff2T, 4096, 1024);

  // 2) LN1
  enc_layernorm<<<8192, blk, 0, stream>>>(x, g1, be1, hbuf);

  // 3) qkv = LN1(x) @ W_qkv + b_qkv            (bf16 out)
  enc_gemm<0><<<dim3(24, 64, 1), blk, 0, stream>>>(
      hbuf, 1024, 0, wqkvT, 1024, 0, qkv, 3072, 0, b_qkv, nullptr, 0, 1024, 1.f);

  // 4) transpose V slice per batch
  enc_vtranspose<<<(4 * 1024 * 2048) / 256, blk, 0, stream>>>(qkv, vt);

  // 5) scores = Q @ K^T / sqrt(d)              (f32 out, batched)
  enc_gemm<1><<<dim3(16, 16, 4), blk, 0, stream>>>(
      qkv, 3072, 2048LL * 3072,            // A = Q slice
      qkv + 1024, 3072, 2048LL * 3072,     // Bt rows = tokens of K slice
      scoresF, 2048, 2048LL * 2048,
      nullptr, nullptr, 0, 1024, 0.03125f);

  // 6) softmax rows -> bf16 P
  enc_softmax<<<8192, blk, 0, stream>>>(scoresF, P);

  // 7) attn_out = P @ V                        (bf16 out, batched)
  enc_gemm<0><<<dim3(8, 16, 4), blk, 0, stream>>>(
      P, 2048, 2048LL * 2048, vt, 2048, 1024LL * 2048,
      attno, 1024, 2048LL * 1024, nullptr, nullptr, 0, 2048, 1.f);

  // 8) x1 = x + attn_out @ W_o + b_o           (f32 into d_out)
  enc_gemm<2><<<dim3(8, 64, 1), blk, 0, stream>>>(
      attno, 1024, 0, woT, 1024, 0, out, 1024, 0, b_o, x, 0, 1024, 1.f);

  // 9) LN2
  enc_layernorm<<<8192, blk, 0, stream>>>(out, g2, be2, hbuf);

  // 10) ff1 = gelu(LN2(x1) @ W_ff1 + b_ff1)    (bf16 out)
  enc_gemm<3><<<dim3(32, 64, 1), blk, 0, stream>>>(
      hbuf, 1024, 0, wff1T, 1024, 0, ff1, 4096, 0, b_ff1, nullptr, 0, 1024, 1.f);

  // 11) out = x1 + ff1 @ W_ff2 + b_ff2         (f32, in-place residual on d_out)
  enc_gemm<2><<<dim3(8, 64, 1), blk, 0, stream>>>(
      ff1, 4096, 0, wff2T, 4096, 0, out, 1024, 0, b_ff2, out, 0, 4096, 1.f);
}
